// EdgeEncoder_59803124629556
// MI455X (gfx1250) — matrix operand order repaired and checked by reading, not compiled, not run on hardware
//
#include <hip/hip_runtime.h>
#include <math.h>

typedef __attribute__((ext_vector_type(16))) __bf16 v16bf;
typedef __attribute__((ext_vector_type(8)))  float  v8f;

#define EPS_ 1e-8f
#define B_   64
#define L_   128
#define D_   1024
#define N_   32
#define OM_  36
#define OE_  36
#define JROWS_ (N_*OE_)   /* 1152 entity-object rows per batch */
#define MPAD_  48         /* 36 mention rows padded to 3x16 tiles */
#define LSTR_  1032       /* LDS row stride (elems): 1024 + 8 pad to kill bank conflicts */

// ---------------- mention span mean-pool: m_enc[b,d] ----------------
__global__ void pool_kernel(const float* __restrict__ mtf, const int* __restrict__ sp,
                            const int* __restrict__ ep, float* __restrict__ m_enc) {
  int b = blockIdx.x;
  int s = sp[b], e = ep[b];
  float inv = 1.0f / (float)(e - s + 1);
  for (int d = threadIdx.x; d < D_; d += blockDim.x) {
    float acc = 0.f;
    for (int l = s; l <= e; ++l) acc += mtf[((size_t)b * L_ + l) * D_ + d];
    m_enc[(size_t)b * D_ + d] = acc * inv;
  }
}

// ---------------- mtet[b,n]: cosine(m_enc, entity_text) ----------------
__global__ void mtet_kernel(const float* __restrict__ m_enc, const float* __restrict__ etf,
                            float* __restrict__ out) {
  int wid = threadIdx.x >> 5, lane = threadIdx.x & 31;
  int pair = blockIdx.x * 8 + wid;               // 2048 (b,n) pairs
  int b = pair >> 5, n = pair & 31;
  const float* mv = m_enc + (size_t)b * D_;
  const float* ev = etf + ((size_t)b * N_ + n) * D_;
  float dot = 0.f, na = 0.f, nb = 0.f;
  for (int d = lane; d < D_; d += 32) {
    float m = mv[d], e = ev[d];
    dot += m * e; na += m * m; nb += e * e;
  }
  for (int off = 16; off > 0; off >>= 1) {
    dot += __shfl_down(dot, off);
    na  += __shfl_down(na,  off);
    nb  += __shfl_down(nb,  off);
  }
  if (lane == 0)
    out[pair] = dot / (fmaxf(sqrtf(na), EPS_) * fmaxf(sqrtf(nb), EPS_));
}

// ---------------- mention-object row norms ----------------
__global__ void mnnorm_kernel(const float* __restrict__ mof, float* __restrict__ mn_norm) {
  int wid = threadIdx.x >> 5, lane = threadIdx.x & 31;
  int row = blockIdx.x * 8 + wid;                // 2304 rows
  const float* rp = mof + (size_t)row * D_;
  float s = 0.f;
  for (int d = lane; d < D_; d += 32) { float v = rp[d]; s += v * v; }
  for (int off = 16; off > 0; off >>= 1) s += __shfl_down(s, off);
  if (lane == 0) mn_norm[row] = sqrtf(s);
}

__global__ void zero_kernel(float* p, int n) {
  int i = blockIdx.x * blockDim.x + threadIdx.x;
  if (i < n) p[i] = 0.f;
}

// ---------------- miei core: bf16 WMMA GEMM + fused entity norms ----------------
__global__ void miei_wmma_kernel(const float* __restrict__ mof, const float* __restrict__ msc,
                                 const float* __restrict__ eof, const float* __restrict__ esc,
                                 const float* __restrict__ mn_norm, float* __restrict__ num_ws) {
  extern __shared__ char smem[];
  __bf16* ldsA = (__bf16*)smem;                               // [MPAD_][LSTR_]
  float*  ldsw = (float*)(smem + (size_t)MPAD_ * LSTR_ * sizeof(__bf16));
  int b = blockIdx.x, tid = threadIdx.x;

  // Stage A (mention objects) as bf16, zero-padded rows 36..47
  for (int idx = tid; idx < MPAD_ * D_; idx += blockDim.x) {
    int r = idx >> 10, k = idx & (D_ - 1);
    float v = (r < OM_) ? mof[((size_t)b * OM_ + r) * D_ + k] : 0.0f;
    ldsA[r * LSTR_ + k] = (__bf16)v;
  }
  if (tid < MPAD_) {
    float w = 0.f;
    if (tid < OM_)
      w = msc[(size_t)b * OM_ + tid] / fmaxf(mn_norm[(size_t)b * OM_ + tid], EPS_);
    ldsw[tid] = w;   // score_m / max(||m||, eps); 0 for pad rows
  }
  __syncthreads();

  int wid = tid >> 5, lane = tid & 31;
  int kh = lane >> 4, nn = lane & 15;   // kh = K-half, nn = column-in-tile / row-in-tile

  for (int t = 0; t < 9; ++t) {
    int c0 = (wid * 9 + t) << 4;                 // column tile base (0..1136)
    int j  = c0 + nn;                            // this lane's entity-object row
    const float* erow = eof + (((size_t)b * JROWS_ + j) << 10);
    v8f acc[3] = {};
    float ss = 0.f;                              // partial sum-of-squares of row j
    for (int k0 = 0; k0 < D_; k0 += 32) {
      // B operand: 16 consecutive f32 of row j (K = k0 + kh*16 .. +15), aligned float4s
      const float4* bp = (const float4*)(erow + k0 + kh * 16);
      float4 q0 = bp[0], q1 = bp[1], q2 = bp[2], q3 = bp[3];
      v16bf bv;
      bv[0]=(__bf16)q0.x; bv[1]=(__bf16)q0.y; bv[2]=(__bf16)q0.z; bv[3]=(__bf16)q0.w;
      bv[4]=(__bf16)q1.x; bv[5]=(__bf16)q1.y; bv[6]=(__bf16)q1.z; bv[7]=(__bf16)q1.w;
      bv[8]=(__bf16)q2.x; bv[9]=(__bf16)q2.y; bv[10]=(__bf16)q2.z; bv[11]=(__bf16)q2.w;
      bv[12]=(__bf16)q3.x; bv[13]=(__bf16)q3.y; bv[14]=(__bf16)q3.z; bv[15]=(__bf16)q3.w;
      ss += q0.x*q0.x + q0.y*q0.y + q0.z*q0.z + q0.w*q0.w
          + q1.x*q1.x + q1.y*q1.y + q1.z*q1.z + q1.w*q1.w
          + q2.x*q2.x + q2.y*q2.y + q2.z*q2.z + q2.w*q2.w
          + q3.x*q3.x + q3.y*q3.y + q3.z*q3.z + q3.w*q3.w;
      // A operands: 16 consecutive K per lane (K = k0 + kh*16 .. +15), matching B
      #pragma unroll
      for (int mt = 0; mt < 3; ++mt) {
        const __bf16* ar = ldsA + (mt * 16 + nn) * LSTR_ + k0 + kh * 16;
        v16bf av;
        #pragma unroll
        for (int i = 0; i < 8; ++i) { av[i] = ar[i]; av[8 + i] = ar[8 + i]; }
        acc[mt] = __builtin_amdgcn_wmma_f32_16x16x32_bf16(
            false, av, false, bv, (short)0, acc[mt], false, false);
      }
    }
    // combine the two lane-halves of row j's sum-of-squares (wave32 xor-16)
    float ssf = ss + __shfl_xor(ss, 16);
    float inv_e = 1.0f / fmaxf(sqrtf(ssf), EPS_);
    float se = esc[(size_t)b * JROWS_ + j];
    int nent = j / OE_;
    // D layout: lane holds N=nn, M = kh*8 + v (+ 16*mt)
    float contrib = 0.f;
    #pragma unroll
    for (int mt = 0; mt < 3; ++mt) {
      #pragma unroll
      for (int v = 0; v < 8; ++v)
        contrib += acc[mt][v] * ldsw[mt * 16 + kh * 8 + v];
    }
    contrib *= inv_e * se;
    atomicAdd(&num_ws[b * N_ + nent], contrib);
  }
}

// ---------------- finalize: miei = num / (den + 1e-9) ----------------
__global__ void final_kernel(const float* __restrict__ msc, const float* __restrict__ esc,
                             const float* __restrict__ num_ws, float* __restrict__ out) {
  int idx = blockIdx.x * blockDim.x + threadIdx.x;
  if (idx >= B_ * N_) return;
  int b = idx >> 5;
  float sm = 0.f;
  for (int m = 0; m < OM_; ++m) sm += msc[b * OM_ + m];
  float se = 0.f;
  for (int o = 0; o < OE_; ++o) se += esc[(size_t)idx * OE_ + o];
  out[B_ * N_ + idx] = num_ws[idx] / (sm * se + 1e-9f);
}

extern "C" void kernel_launch(void* const* d_in, const int* in_sizes, int n_in,
                              void* d_out, int out_size, void* d_ws, size_t ws_size,
                              hipStream_t stream) {
  (void)in_sizes; (void)n_in; (void)out_size; (void)ws_size;
  const float* mtf = (const float*)d_in[0];   // [64,128,1024]
  const int*   sp  = (const int*)  d_in[1];   // [64]
  const int*   ep  = (const int*)  d_in[2];   // [64]
  const float* mof = (const float*)d_in[3];   // [64,36,1024]
  const float* msc = (const float*)d_in[4];   // [64,36]
  const float* etf = (const float*)d_in[5];   // [64,32,1024]
  const float* eof = (const float*)d_in[6];   // [64,32,36,1024]
  const float* esc = (const float*)d_in[7];   // [64,32,36]
  float* out = (float*)d_out;                 // [2048 mtet | 2048 miei]

  float* ws      = (float*)d_ws;
  float* m_enc   = ws;                        // 64*1024
  float* mn_norm = ws + (size_t)B_ * D_;      // 64*36
  float* num_ws  = mn_norm + B_ * OM_;        // 64*32

  pool_kernel  <<<B_, 256, 0, stream>>>(mtf, sp, ep, m_enc);
  mtet_kernel  <<<(B_ * N_) / 8, 256, 0, stream>>>(m_enc, etf, out);
  mnnorm_kernel<<<(B_ * OM_) / 8, 256, 0, stream>>>(mof, mn_norm);
  zero_kernel  <<<(B_ * N_ + 255) / 256, 256, 0, stream>>>(num_ws, B_ * N_);

  size_t lds_bytes = (size_t)MPAD_ * LSTR_ * sizeof(__bf16) + MPAD_ * sizeof(float);
  miei_wmma_kernel<<<B_, 256, lds_bytes, stream>>>(mof, msc, eof, esc, mn_norm, num_ws);

  final_kernel <<<(B_ * N_ + 255) / 256, 256, 0, stream>>>(msc, esc, num_ws, out);
}
